// LinearDecoder_19387482374234
// MI455X (gfx1250) — matrix operand order repaired and checked by reading, not compiled, not run on hardware
//
#include <hip/hip_runtime.h>
#include <math.h>

typedef _Float16 v16h __attribute__((ext_vector_type(16)));
typedef float    v8f  __attribute__((ext_vector_type(8)));
typedef uint32_t v4u  __attribute__((ext_vector_type(4)));

#define BDIM 131072
#define XK   128      // K
#define YN   512      // N total
#define BM   128      // rows per block
#define BN   64       // cols per block
#define SA   136      // padded LDS stride (halves) for A/E rows  [BM][SA]
#define SW   72       // Wt/W2 region size (XK*SW halves each); rows are [BN][SA]
#define NTIL 4        // 16-col tiles per wave (BN/16)
#define KSTP 4        // K steps (XK/32)

// Byte layout of the dynamic-LDS arena:
//   smA  : [BM][SA] f16   at 0            (34816 B)
//   smE  : [BM][SA] f16   at OFF_E        (34816 B)
//   smWt : [BN][SA] f16   at OFF_W        (18432 B)
//   smW2 : [BN][SA] f16   at OFF_W + OFF_W2
//   smb  : [BN] f32
#define ABYTES  (BM * SA * 2)     // 34816
#define WBYTES  (XK * SW * 2)     // 18432
#define OFF_E   ABYTES
#define OFF_W   (2 * ABYTES)
#define OFF_W2  WBYTES

// A-matrix 16x32 f16 fragment (ISA 7.12.2): lane=(m=lane&15, half=lane>>4).
// From row base (+half*16 folded into base): two b128 runs at +ks*64 and +ks*64+32.
__device__ __forceinline__ v16h ld_fragA(const char* base, int off) {
  union { v16h v; v4u q[2]; } r;
  r.q[0] = *(const v4u*)(base + off);
  r.q[1] = *(const v4u*)(base + off + 32);
  return r.v;
}

// B-matrix 32x16 f16 fragment: one contiguous 32 B run -> 2x b128. Unused:
// B comes from [n][k] Wt rows (lane <-> N, halves <-> K) via ld_fragA.
__device__ __forceinline__ v16h ld_fragB(const char* base, int off) {
  union { v16h v; v4u q[2]; } r;
  r.q[0] = *(const v4u*)(base + off);
  r.q[1] = *(const v4u*)(base + off + 16);
  return r.v;
}

extern "C" __global__ __launch_bounds__(256)
void LinearDecoder_19387482374234_kernel(const float* __restrict__ mean,
                                         const float* __restrict__ logvar,
                                         const float* __restrict__ W,
                                         const float* __restrict__ b,
                                         float* __restrict__ out) {
  extern __shared__ char smem[];
  _Float16* smA  = (_Float16*)smem;
  _Float16* smE  = (_Float16*)(smem + OFF_E);
  _Float16* smWt = (_Float16*)(smem + OFF_W);
  _Float16* smW2 = (_Float16*)(smem + OFF_W + OFF_W2);
  float*    smb  = (float*)(smem + OFF_W + 2 * WBYTES);

  const int tid = threadIdx.x;
  const int nb0 = blockIdx.x * BN;
  const int mb0 = blockIdx.y * BM;

  // ---- stage W slice as [n][k] (stride SA) f16 (stays L2-resident) ----
  for (int i = tid; i < BN * XK; i += 256) {
    const int n = i >> 7, k = i & (XK - 1);     // consecutive tid -> consecutive k
    const float w = W[(size_t)(nb0 + n) * XK + k];
    smWt[n * SA + k] = (_Float16)w;
    smW2[n * SA + k] = (_Float16)(w * w);
  }
  if (tid < BN) smb[tid] = b[nb0 + tid];

  // ---- stage mean -> f16, exp(logvar) -> f16; float4 loads, b64 LDS stores ----
  const float4* m4 = (const float4*)(mean   + (size_t)mb0 * XK);
  const float4* l4 = (const float4*)(logvar + (size_t)mb0 * XK);
  for (int i = tid; i < BM * XK / 4; i += 256) {
    const int r = i >> 5;              // 32 float4 per row
    const int c = (i & 31) << 2;
    union { _Float16 h[4]; uint2 u; } pk;
    const float4 vm = m4[i];
    pk.h[0] = (_Float16)vm.x; pk.h[1] = (_Float16)vm.y;
    pk.h[2] = (_Float16)vm.z; pk.h[3] = (_Float16)vm.w;
    *(uint2*)(smA + r * SA + c) = pk.u;
    const float4 vl = l4[i];
    pk.h[0] = (_Float16)__expf(vl.x); pk.h[1] = (_Float16)__expf(vl.y);
    pk.h[2] = (_Float16)__expf(vl.z); pk.h[3] = (_Float16)__expf(vl.w);
    *(uint2*)(smE + r * SA + c) = pk.u;
  }
  __syncthreads();

  const int wave = tid >> 5;
  const int lane = tid & 31;
  const int lm   = lane & 15;
  const int half = lane >> 4;

  // Lane-dependent bases; everything else is a compile-time immediate offset
  // that fits the 16-bit DS offset field (max ~35 KB from each base).
  const char* baseA = smem + (wave * 16 + lm) * (SA * 2) + half * 16;
  const char* baseW = smem + OFF_W + lm * (SA * 2) + half * 16;

  const v8f zero = {0.f, 0.f, 0.f, 0.f, 0.f, 0.f, 0.f, 0.f};
  v8f accM[NTIL], accV[NTIL];
#pragma unroll
  for (int t = 0; t < NTIL; ++t) { accM[t] = zero; accV[t] = zero; }

#pragma unroll
  for (int ks = 0; ks < KSTP; ++ks) {
    const v16h aM = ld_fragA(baseA, ks * 64);
    const v16h aE = ld_fragA(baseA, OFF_E + ks * 64);
#pragma unroll
    for (int t = 0; t < NTIL; ++t) {
      const v16h bW  = ld_fragA(baseW, ks * 64 + t * (16 * SA * 2));
      const v16h bW2 = ld_fragA(baseW, OFF_W2 + ks * 64 + t * (16 * SA * 2));
      accM[t] = __builtin_amdgcn_wmma_f32_16x16x32_f16(
          false, aM, false, bW,  (short)0, accM[t], false, false);
      accV[t] = __builtin_amdgcn_wmma_f32_16x16x32_f16(
          false, aE, false, bW2, (short)0, accV[t], false, false);
    }
  }

  // ---- epilogue: C/D layout: N = lane&15, M = half*8 + r (ISA 7.12.2) ----
  const size_t mrow = (size_t)mb0 + (size_t)(wave * 16 + half * 8);
  float* outM = out + mrow * YN + nb0;
  float* outV = out + (size_t)BDIM * YN + mrow * YN + nb0;
#pragma unroll
  for (int t = 0; t < NTIL; ++t) {
    const int n = t * 16 + lm;
    const float bias = smb[n];
#pragma unroll
    for (int r = 0; r < 8; ++r) {
      outM[(size_t)r * YN + n] = accM[t][r] + bias;
      outV[(size_t)r * YN + n] = __logf(accV[t][r]);
    }
  }
}

extern "C" void kernel_launch(void* const* d_in, const int* in_sizes, int n_in,
                              void* d_out, int out_size, void* d_ws, size_t ws_size,
                              hipStream_t stream) {
  (void)in_sizes; (void)n_in; (void)out_size; (void)d_ws; (void)ws_size;
  const float* mean   = (const float*)d_in[0];
  const float* logvar = (const float*)d_in[1];
  const float* W      = (const float*)d_in[2];
  const float* b      = (const float*)d_in[3];
  float* out = (float*)d_out;

  dim3 grid(YN / BN, BDIM / BM);   // (8, 1024)
  dim3 block(256);                  // 8 wave32 waves
  const size_t shmem = (size_t)OFF_W + 2 * WBYTES + BN * sizeof(float);
  hipLaunchKernelGGL(LinearDecoder_19387482374234_kernel, grid, block, shmem, stream,
                     mean, logvar, W, b, out);
}
